// SNN_43439299231835
// MI455X (gfx1250) — matrix operand order repaired and checked by reading, not compiled, not run on hardware
//
#include <hip/hip_runtime.h>
#include <hip/hip_bf16.h>
#include <cstdint>

#define ALPHA_MIN 0.8187307530779818f
#define ALPHA_MAX 0.9607894391523232f

typedef __attribute__((ext_vector_type(16))) __bf16 v16bf;
typedef __attribute__((ext_vector_type(8)))  float  v8f;
typedef __attribute__((ext_vector_type(4)))  unsigned int v4u;

union Frag { v4u u[2]; v16bf v; };

__device__ __forceinline__ v8f wmma_bf16(v16bf a, v16bf b, v8f c) {
  // 8 args: (neg_a, A, neg_b, B, c_mod, C, reuse_a, reuse_b)
  return __builtin_amdgcn_wmma_f32_16x16x32_bf16(false, a, false, b, (short)0, c,
                                                 false, false);
}

// ---------------------------------------------------------------------------
// GEMM: C[M,N] = A[M,Kp] * W[N,Kp]^T   (bf16 in, f32 out)
// Block = 128 threads = 4 waves; each wave computes a 32x64 C tile.
// A-frag per lane: row m = m0+l15, chunks at k0+hi*8 and k0+16+hi*8 (b128 each)
// B-frag per lane: col n = n0+l15, chunks at k0+hi*8 and k0+16+hi*8 (b128 each)
// ---------------------------------------------------------------------------
__global__ void __launch_bounds__(128)
k_gemm_bf16(const __bf16* __restrict__ A, const __bf16* __restrict__ Bm,
            float* __restrict__ C, int Kp, int N) {
  const int tid  = threadIdx.x;
  const int wave = tid >> 5;
  const int lane = tid & 31;
  const int l15  = lane & 15, hi = lane >> 4;
  const int n0w  = blockIdx.x * 128 + (wave & 1) * 64;
  const int m0w  = blockIdx.y * 64  + (wave >> 1) * 32;

  const v8f zero = {0.f, 0.f, 0.f, 0.f, 0.f, 0.f, 0.f, 0.f};
  v8f acc[2][4];
  #pragma unroll
  for (int i = 0; i < 2; ++i)
    #pragma unroll
    for (int j = 0; j < 4; ++j) acc[i][j] = zero;

  const int nK = Kp >> 5;
  for (int ks = 0; ks < nK; ++ks) {
    const int k0 = ks << 5;
    Frag fa[2], fb[4];
    #pragma unroll
    for (int mt = 0; mt < 2; ++mt) {
      const __bf16* pa = A + (size_t)(m0w + mt * 16 + l15) * Kp + k0 + hi * 8;
      fa[mt].u[0] = *(const v4u*)pa;
      fa[mt].u[1] = *(const v4u*)(pa + 16);
    }
    #pragma unroll
    for (int nt = 0; nt < 4; ++nt) {
      const __bf16* pb = Bm + (size_t)(n0w + nt * 16 + l15) * Kp + k0 + hi * 8;
      fb[nt].u[0] = *(const v4u*)pb;
      fb[nt].u[1] = *(const v4u*)(pb + 16);
    }
    #pragma unroll
    for (int mt = 0; mt < 2; ++mt)
      #pragma unroll
      for (int nt = 0; nt < 4; ++nt)
        acc[mt][nt] = wmma_bf16(fa[mt].v, fb[nt].v, acc[mt][nt]);
  }

  #pragma unroll
  for (int mt = 0; mt < 2; ++mt)
    #pragma unroll
    for (int nt = 0; nt < 4; ++nt)
      #pragma unroll
      for (int r = 0; r < 8; ++r) {
        const int m = m0w + mt * 16 + r + hi * 8;
        const int n = n0w + nt * 16 + l15;
        C[(size_t)m * N + n] = acc[mt][nt][r];
      }
}

// ---------------------------------------------------------------------------
// Grid barrier (persistent scan kernel, 32 WGs)
// ---------------------------------------------------------------------------
__device__ __forceinline__ void grid_wait(unsigned* cnt, unsigned target) {
  __syncthreads();
  if (threadIdx.x == 0) {
    __hip_atomic_fetch_add(cnt, 1u, __ATOMIC_RELEASE, __HIP_MEMORY_SCOPE_AGENT);
    while (__hip_atomic_load(cnt, __ATOMIC_ACQUIRE, __HIP_MEMORY_SCOPE_AGENT) < target)
      __builtin_amdgcn_s_sleep(2);
  }
  __syncthreads();
}

// ---------------------------------------------------------------------------
// RLIF scan layer.  32 WGs x 64 threads (2 waves). WG owns 16 columns of H.
//   VzT slice (512x16 bf16 = 16KB) resident in LDS for the whole scan.
//   ut kept in registers in WMMA C-layout; spikes double-buffered bf16 global.
//   rec = st_prev @ Vz via 2 waves * 16 k-steps of v_wmma_f32_16x16x32_bf16.
// ---------------------------------------------------------------------------
__global__ void __launch_bounds__(64)
k_scan(const float* __restrict__ Wx,      // [32,1000,512] f32
       const __bf16* __restrict__ VzT,    // [512,512]: row g = column g of Vz
       const float* __restrict__ alpha,
       const float* __restrict__ ut0, const float* __restrict__ st0,
       __bf16* __restrict__ sout,         // [32,1000,512]
       __bf16* __restrict__ stbuf,        // [2][32*512]
       unsigned* __restrict__ cnt) {
  __shared__ __bf16 ldsB[16 * 512];
  const int g0   = blockIdx.x * 16;
  const int tid  = threadIdx.x;
  const int wave = tid >> 5;
  const int lane = tid & 31;
  const int l15  = lane & 15, hi = lane >> 4;
  const int bbase = wave * 16;   // wave0: b=0..15, wave1: b=16..31
  const int g     = g0 + l15;

  // stage VzT rows g0..g0+15 (contiguous) into LDS, b128 per thread-iter
  for (int i = tid; i < (16 * 512) / 8; i += 64)
    ((v4u*)ldsB)[i] = ((const v4u*)(VzT + (size_t)g0 * 512))[i];
  __syncthreads();

  const float ag = fminf(fmaxf(alpha[g], ALPHA_MIN), ALPHA_MAX);
  float ut[8], st[8];
  #pragma unroll
  for (int r = 0; r < 8; ++r) {
    const int b = bbase + r + hi * 8;
    ut[r] = ut0[b * 512 + g];
    st[r] = st0[b * 512 + g];
    stbuf[b * 512 + g] = (__bf16)st[r];     // buffer 0 holds st at t=0
  }
  grid_wait(cnt, 32u);

  for (int t = 0; t < 1000; ++t) {
    const __bf16* stp = stbuf + (size_t)(t & 1) * (32 * 512);
    __bf16*       stc = stbuf + (size_t)((t + 1) & 1) * (32 * 512);

    v8f rec = {0.f, 0.f, 0.f, 0.f, 0.f, 0.f, 0.f, 0.f};
    const __bf16* pabase = stp + (size_t)(bbase + l15) * 512 + hi * 8;
    #pragma unroll
    for (int ks = 0; ks < 16; ++ks) {
      Frag fa, fb;
      const __bf16* pa = pabase + ks * 32;
      fa.u[0] = *(const v4u*)pa;
      fa.u[1] = *(const v4u*)(pa + 16);
      const __bf16* pb = ldsB + l15 * 512 + ks * 32 + hi * 8;
      fb.u[0] = *(const v4u*)pb;
      fb.u[1] = *(const v4u*)(pb + 16);
      rec = wmma_bf16(fa.v, fb.v, rec);
    }

    #pragma unroll
    for (int r = 0; r < 8; ++r) {
      const int b = bbase + r + hi * 8;
      const float wx = Wx[((size_t)b * 1000 + t) * 512 + g];
      const float u  = ag * (ut[r] - st[r]) + (1.0f - ag) * (wx + rec[r]);
      const float s  = (u > 1.0f) ? 1.0f : 0.0f;
      ut[r] = u; st[r] = s;
      const __bf16 sb = (__bf16)s;
      stc[b * 512 + g] = sb;
      sout[((size_t)b * 1000 + t) * 512 + g] = sb;
    }
    grid_wait(cnt, 32u * (t + 2));
  }
}

// ---------------------------------------------------------------------------
// Readout: leaky integrate + per-(b,t) softmax over H, summed over t.
// One block per batch element, 512 threads.
// ---------------------------------------------------------------------------
__global__ void __launch_bounds__(512)
k_readout(const float* __restrict__ Wx, const float* __restrict__ alphar,
          const float* __restrict__ ut0r, float* __restrict__ out) {
  __shared__ float red[16];
  __shared__ float bc;
  const int b = blockIdx.x, h = threadIdx.x;
  const int wid = h >> 5, lane = h & 31;
  const float a = fminf(fmaxf(alphar[h], ALPHA_MIN), ALPHA_MAX);
  float ut  = ut0r[b * 512 + h];
  float acc = 0.0f;
  const float* wp = Wx + (size_t)b * 1000 * 512 + h;
  for (int t = 0; t < 1000; ++t) {
    ut = a * ut + (1.0f - a) * wp[(size_t)t * 512];
    float m = ut;
    #pragma unroll
    for (int o = 16; o > 0; o >>= 1) m = fmaxf(m, __shfl_xor(m, o, 32));
    if (lane == 0) red[wid] = m;
    __syncthreads();
    if (h == 0) {
      float mm = red[0];
      for (int i = 1; i < 16; ++i) mm = fmaxf(mm, red[i]);
      bc = mm;
    }
    __syncthreads();
    const float e = __expf(ut - bc);
    float s = e;
    #pragma unroll
    for (int o = 16; o > 0; o >>= 1) s += __shfl_xor(s, o, 32);
    if (lane == 0) red[wid] = s;
    __syncthreads();
    if (h == 0) {
      float ss = red[0];
      for (int i = 1; i < 16; ++i) ss += red[i];
      bc = ss;
    }
    __syncthreads();
    acc += e / bc;
  }
  out[b * 512 + h] = acc;
}

// ---------------------------------------------------------------------------
// Firing rates: mean over (b,t) of concat(s0,s1) -> 1024 values
// ---------------------------------------------------------------------------
__global__ void __launch_bounds__(256)
k_rates(const __bf16* __restrict__ s0, const __bf16* __restrict__ s1,
        float* __restrict__ fr) {
  __shared__ float red[256];
  const int j = blockIdx.x;                 // 0..1023
  const __bf16* s = (j < 512) ? s0 : s1;
  const int c = j & 511;
  float acc = 0.f;
  for (int i = threadIdx.x; i < 32000; i += 256)
    acc += (float)s[(size_t)i * 512 + c];
  red[threadIdx.x] = acc;
  __syncthreads();
  for (int o = 128; o > 0; o >>= 1) {
    if (threadIdx.x < o) red[threadIdx.x] += red[threadIdx.x + o];
    __syncthreads();
  }
  if (threadIdx.x == 0) fr[j] = red[0] * (1.0f / 32000.0f);
}

// ---------------------------------------------------------------------------
// Converters
// ---------------------------------------------------------------------------
__global__ void k_cvt_pad(const float* __restrict__ src, __bf16* __restrict__ dst,
                          int rows, int K, int Kp) {
  const size_t total = (size_t)rows * Kp;
  for (size_t idx = (size_t)blockIdx.x * blockDim.x + threadIdx.x; idx < total;
       idx += (size_t)gridDim.x * blockDim.x) {
    const size_t r = idx / Kp;
    const int j = (int)(idx - r * Kp);
    dst[idx] = (j < K) ? (__bf16)src[r * K + j] : (__bf16)0.0f;
  }
}

__global__ void k_vzt(const float* __restrict__ V, __bf16* __restrict__ VzT) {
  const int idx = blockIdx.x * blockDim.x + threadIdx.x;  // 512*512
  if (idx < 512 * 512) {
    const int gg = idx >> 9, hh = idx & 511;
    VzT[idx] = (gg == hh) ? (__bf16)0.0f : (__bf16)V[hh * 512 + gg];
  }
}

__global__ void k_init(unsigned* cnt) { cnt[threadIdx.x] = 0u; }

// ---------------------------------------------------------------------------
extern "C" void kernel_launch(void* const* d_in, const int* in_sizes, int n_in,
                              void* d_out, int out_size, void* d_ws, size_t ws_size,
                              hipStream_t stream) {
  (void)in_sizes; (void)n_in; (void)out_size; (void)ws_size;
  const float* x      = (const float*)d_in[0];
  const float* W0     = (const float*)d_in[1];
  const float* V0     = (const float*)d_in[2];
  const float* alpha0 = (const float*)d_in[3];
  const float* W1     = (const float*)d_in[4];
  const float* V1     = (const float*)d_in[5];
  const float* alpha1 = (const float*)d_in[6];
  const float* Wr     = (const float*)d_in[7];
  const float* alphar = (const float*)d_in[8];
  const float* ut0_0  = (const float*)d_in[9];
  const float* st0_0  = (const float*)d_in[10];
  const float* ut0_1  = (const float*)d_in[11];
  const float* st0_1  = (const float*)d_in[12];
  const float* ut0_r  = (const float*)d_in[13];
  float* out = (float*)d_out;

  char* ws = (char*)d_ws;
  size_t off = 0;
  auto alloc = [&](size_t bytes) {
    char* p = ws + off;
    off = (off + bytes + 255) & ~(size_t)255;
    return p;
  };
  __bf16* xb    = (__bf16*)alloc(32000ull * 704 * 2);
  __bf16* W0b   = (__bf16*)alloc(512ull * 704 * 2);
  __bf16* W1b   = (__bf16*)alloc(512ull * 512 * 2);
  __bf16* Wrb   = (__bf16*)alloc(512ull * 512 * 2);
  __bf16* V0T   = (__bf16*)alloc(512ull * 512 * 2);
  __bf16* V1T   = (__bf16*)alloc(512ull * 512 * 2);
  float*  Wx    = (float*) alloc(32000ull * 512 * 4);
  __bf16* s0b   = (__bf16*)alloc(32000ull * 512 * 2);
  __bf16* s1b   = (__bf16*)alloc(32000ull * 512 * 2);
  __bf16* stbuf = (__bf16*)alloc(2ull * 32 * 512 * 2);
  unsigned* cnt = (unsigned*)alloc(256);

  k_init<<<1, 2, 0, stream>>>(cnt);
  k_cvt_pad<<<2048, 256, 0, stream>>>(x,  xb,  32000, 700, 704);
  k_cvt_pad<<<512,  256, 0, stream>>>(W0, W0b, 512,   700, 704);
  k_cvt_pad<<<512,  256, 0, stream>>>(W1, W1b, 512,   512, 512);
  k_cvt_pad<<<512,  256, 0, stream>>>(Wr, Wrb, 512,   512, 512);
  k_vzt<<<1024, 256, 0, stream>>>(V0, V0T);
  k_vzt<<<1024, 256, 0, stream>>>(V1, V1T);

  const dim3 gg(512 / 128, 32000 / 64);
  k_gemm_bf16<<<gg, 128, 0, stream>>>(xb, W0b, Wx, 704, 512);
  k_scan<<<32, 64, 0, stream>>>(Wx, V0T, alpha0, ut0_0, st0_0, s0b, stbuf, cnt + 0);
  k_gemm_bf16<<<gg, 128, 0, stream>>>(s0b, W1b, Wx, 512, 512);
  k_scan<<<32, 64, 0, stream>>>(Wx, V1T, alpha1, ut0_1, st0_1, s1b, stbuf, cnt + 1);
  k_gemm_bf16<<<gg, 128, 0, stream>>>(s1b, Wrb, Wx, 512, 512);
  k_readout<<<32, 512, 0, stream>>>(Wx, alphar, ut0_r, out);
  k_rates<<<1024, 256, 0, stream>>>(s0b, s1b, out + 32 * 512);
}
